// QuantizedLinear_46849503265229
// MI455X (gfx1250) — matrix operand order repaired and checked by reading, not compiled, not run on hardware
//
#include <hip/hip_runtime.h>

typedef __bf16 bf16_t;
typedef __attribute__((ext_vector_type(16))) bf16_t      v16bf;
typedef __attribute__((ext_vector_type(8)))  bf16_t      v8bf;
typedef __attribute__((ext_vector_type(8)))  float       v8f;
typedef __attribute__((ext_vector_type(8)))  unsigned    v8u;

struct alignas(16) U128 { unsigned u[4]; };

#if defined(__AMDGCN__) && __has_builtin(__builtin_amdgcn_global_load_async_to_lds_b128)
#define USE_ASYNC_LDS 1
typedef int v4i_ __attribute__((vector_size(16)));
typedef __attribute__((address_space(1))) v4i_ v4i_gas;   // global (AS1)
typedef __attribute__((address_space(3))) v4i_ v4i_lds;   // LDS (AS3)
#else
#define USE_ASYNC_LDS 0
#endif

#if defined(__AMDGCN__) && __has_builtin(__builtin_amdgcn_s_wait_asynccnt)
#define WAIT_ASYNC(n) __builtin_amdgcn_s_wait_asynccnt(n)
#else
#define WAIT_ASYNC(n) asm volatile("s_wait_asynccnt %0" ::"n"(n) : "memory")
#endif

// ---------------------------------------------------------------------------
// x (fp32) -> bf16, 8 elements / thread
// ---------------------------------------------------------------------------
__global__ __launch_bounds__(256) void cvt_x_kernel(const float* __restrict__ x,
                                                    bf16_t* __restrict__ xb,
                                                    long long n8) {
  long long i = (long long)blockIdx.x * blockDim.x + threadIdx.x;
  if (i >= n8) return;
  const float* p = x + i * 8;
  v8bf o;
#pragma unroll
  for (int j = 0; j < 8; ++j) o[j] = (bf16_t)p[j];
  *(U128*)(xb + i * 8) = __builtin_bit_cast(U128, o);
}

// ---------------------------------------------------------------------------
// q (int32, 0..255) -> bf16 dequantized weight: w = scale*(q - zp)
// ---------------------------------------------------------------------------
__global__ __launch_bounds__(256) void cvt_w_kernel(const int* __restrict__ q,
                                                    const float* __restrict__ sp,
                                                    const float* __restrict__ zpp,
                                                    bf16_t* __restrict__ wb,
                                                    long long n8) {
  long long i = (long long)blockIdx.x * blockDim.x + threadIdx.x;
  if (i >= n8) return;
  float s = *sp;
  float zs = (*zpp) * s;
  const int* p = q + i * 8;
  v8bf o;
#pragma unroll
  for (int j = 0; j < 8; ++j) o[j] = (bf16_t)((float)p[j] * s - zs);
  *(U128*)(wb + i * 8) = __builtin_bit_cast(U128, o);
}

// ---------------------------------------------------------------------------
// bf16 GEMM: out[m,n] = sum_k Xb[m,k]*Wb[n,k] + bias[n]
// 256 threads = 8 waves (wave32), tile 128x128, BK=64, double-buffered LDS.
// Wave grid 2(M) x 4(N); each wave: 64x32 = 4x2 WMMA 16x16 accumulators.
// ---------------------------------------------------------------------------
static constexpr int BM  = 128;
static constexpr int BN  = 128;
static constexpr int BK  = 64;
static constexpr int LDT = 72;   // padded row stride (bf16 elems) = 144 B

__device__ __forceinline__ void compute_tile(const bf16_t* At, const bf16_t* Bt,
                                             v8f (&acc)[4][2],
                                             int wm, int wn, int hi, int l16) {
#pragma unroll
  for (int kk = 0; kk < BK; kk += 32) {
    v16bf afrag[4];
    v16bf bfrag[2];
#pragma unroll
    for (int sm = 0; sm < 4; ++sm) {
      // A 16x32: lane<16 holds K{0..7,16..23}; lane>=16 shifted by 8 elems
      const bf16_t* p = At + (wm * 64 + sm * 16 + l16) * LDT + kk + hi * 8;
      U128 lo = *(const U128*)p;
      U128 hh = *(const U128*)(p + 16);
      v8u u = {lo.u[0], lo.u[1], lo.u[2], lo.u[3],
               hh.u[0], hh.u[1], hh.u[2], hh.u[3]};
      afrag[sm] = __builtin_bit_cast(v16bf, u);
    }
#pragma unroll
    for (int sn = 0; sn < 2; ++sn) {
      // B 32x16: lane<16 holds K{0..7,16..23}; lane>=16 shifted by 8 elems
      const bf16_t* p = Bt + (wn * 32 + sn * 16 + l16) * LDT + kk + hi * 8;
      U128 q0 = *(const U128*)p;
      U128 q1 = *(const U128*)(p + 16);
      v8u u = {q0.u[0], q0.u[1], q0.u[2], q0.u[3],
               q1.u[0], q1.u[1], q1.u[2], q1.u[3]};
      bfrag[sn] = __builtin_bit_cast(v16bf, u);
    }
#pragma unroll
    for (int sm = 0; sm < 4; ++sm)
#pragma unroll
      for (int sn = 0; sn < 2; ++sn)
        acc[sm][sn] = __builtin_amdgcn_wmma_f32_16x16x32_bf16(
            false, afrag[sm], false, bfrag[sn], (short)0, acc[sm][sn],
            false, false);
  }
}

__global__ __launch_bounds__(256) void gemm_bf16_kernel(
    const bf16_t* __restrict__ Xb, const bf16_t* __restrict__ Wb,
    const float* __restrict__ bias, float* __restrict__ out,
    int M, int N, int K) {
  __shared__ __attribute__((aligned(16))) bf16_t Atile[2][BM * LDT];
  __shared__ __attribute__((aligned(16))) bf16_t Btile[2][BN * LDT];

  const int tid  = threadIdx.x;
  const int lane = tid & 31;
  const int wave = tid >> 5;
  const int wm   = wave & 1;    // 0..1 -> 64 rows each
  const int wn   = wave >> 1;   // 0..3 -> 32 cols each
  const int hi   = lane >> 4;   // half-wave select
  const int l16  = lane & 15;

  const int block_m = blockIdx.y * BM;
  const int block_n = blockIdx.x * BN;

  // global->LDS fill mapping: one 16B chunk / thread / row-group
  const int ldr = tid >> 2;          // 0..63
  const int ldc = (tid & 3) * 8;     // element column (8 bf16 = 16 B)

  v8f acc[4][2];
#pragma unroll
  for (int i = 0; i < 4; ++i)
#pragma unroll
    for (int j = 0; j < 2; ++j) acc[i][j] = (v8f){0, 0, 0, 0, 0, 0, 0, 0};

  const int nk = K / BK;

#if USE_ASYNC_LDS
  // ---- double-buffered pipeline with async global->LDS copies ----
  auto issue = [&](int k0, int buf) {
#pragma unroll
    for (int i = 0; i < 2; ++i) {
      const int row = ldr + i * 64;
      const bf16_t* ga = Xb + (size_t)(block_m + row) * K + k0 + ldc;
      const bf16_t* gb = Wb + (size_t)(block_n + row) * K + k0 + ldc;
      __builtin_amdgcn_global_load_async_to_lds_b128(
          (v4i_gas*)ga, (v4i_lds*)&Atile[buf][row * LDT + ldc], 0, 0);
      __builtin_amdgcn_global_load_async_to_lds_b128(
          (v4i_gas*)gb, (v4i_lds*)&Btile[buf][row * LDT + ldc], 0, 0);
    }
  };
  issue(0, 0);
  for (int i = 0; i < nk; ++i) {
    const int buf = i & 1;
    if (i + 1 < nk) {
      issue((i + 1) * BK, buf ^ 1);   // overlaps with compute below
      WAIT_ASYNC(4);                  // 4 oldest (stage i) retired, in-order
    } else {
      WAIT_ASYNC(0);
    }
    __syncthreads();                  // all waves' stage-i data in LDS
    compute_tile(&Atile[buf][0], &Btile[buf][0], acc, wm, wn, hi, l16);
    __syncthreads();                  // done reading buf before reuse
  }
#else
  // ---- fallback: register-prefetch software pipeline, single LDS buffer ----
  U128 pa[2], pb[2];
  auto fetch = [&](int k0) {
#pragma unroll
    for (int i = 0; i < 2; ++i) {
      const int row = ldr + i * 64;
      pa[i] = *(const U128*)(Xb + (size_t)(block_m + row) * K + k0 + ldc);
      pb[i] = *(const U128*)(Wb + (size_t)(block_n + row) * K + k0 + ldc);
    }
  };
  fetch(0);
  for (int i = 0; i < nk; ++i) {
    __syncthreads();                  // previous compute done reading
#pragma unroll
    for (int j = 0; j < 2; ++j) {
      const int row = ldr + j * 64;
      *(U128*)&Atile[0][row * LDT + ldc] = pa[j];
      *(U128*)&Btile[0][row * LDT + ldc] = pb[j];
    }
    __syncthreads();
    if (i + 1 < nk) fetch((i + 1) * BK);  // global loads fly during WMMAs
    compute_tile(&Atile[0][0], &Btile[0][0], acc, wm, wn, hi, l16);
  }
#endif

  // epilogue: C/D layout -> row = sm*16 + r + 8*hi, col = lane&15
#pragma unroll
  for (int sn = 0; sn < 2; ++sn) {
    const int col = block_n + wn * 32 + sn * 16 + l16;
    const float bv = bias[col];
#pragma unroll
    for (int sm = 0; sm < 4; ++sm) {
      const int row0 = block_m + wm * 64 + sm * 16 + hi * 8;
#pragma unroll
      for (int r = 0; r < 8; ++r)
        out[(size_t)(row0 + r) * N + col] = acc[sm][sn][r] + bv;
    }
  }
}

// ---------------------------------------------------------------------------
// Fallback (workspace too small or M/N/K not tile multiples): naive dequant-on-the-fly
// ---------------------------------------------------------------------------
__global__ __launch_bounds__(256) void naive_kernel(
    const float* __restrict__ x, const int* __restrict__ q,
    const float* __restrict__ sp, const float* __restrict__ zpp,
    const float* __restrict__ bias, float* __restrict__ out,
    int M, int N, int K) {
  long long idx = (long long)blockIdx.x * blockDim.x + threadIdx.x;
  if (idx >= (long long)M * N) return;
  const int m = (int)(idx / N);
  const int n = (int)(idx % N);
  const float s = *sp, z = *zpp;
  float a = 0.f;
  for (int k = 0; k < K; ++k)
    a += x[(size_t)m * K + k] * (s * ((float)q[(size_t)n * K + k] - z));
  out[idx] = a + bias[n];
}

extern "C" void kernel_launch(void* const* d_in, const int* in_sizes, int n_in,
                              void* d_out, int out_size, void* d_ws, size_t ws_size,
                              hipStream_t stream) {
  const float* x     = (const float*)d_in[0];
  const int*   qw    = (const int*)d_in[1];
  const float* scale = (const float*)d_in[2];
  const float* zp    = (const float*)d_in[3];
  const float* bias  = (const float*)d_in[4];
  float*       out   = (float*)d_out;

  const long long xsz = in_sizes[0];
  const long long wsz = in_sizes[1];
  const int N = in_sizes[4];
  const int K = (int)(wsz / N);
  const int M = (int)(xsz / K);

  const size_t needed = (size_t)M * K * 2 + (size_t)N * K * 2;
  if (ws_size < needed || (M % BM) || (N % BN) || (K % BK)) {
    long long total = (long long)M * N;
    naive_kernel<<<dim3((unsigned)((total + 255) / 256)), 256, 0, stream>>>(
        x, qw, scale, zp, bias, out, M, N, K);
    return;
  }

  bf16_t* xb = (bf16_t*)d_ws;
  bf16_t* wb = xb + (size_t)M * K;

  {
    long long n8 = xsz / 8;
    cvt_x_kernel<<<dim3((unsigned)((n8 + 255) / 256)), 256, 0, stream>>>(x, xb, n8);
  }
  {
    long long n8 = wsz / 8;
    cvt_w_kernel<<<dim3((unsigned)((n8 + 255) / 256)), 256, 0, stream>>>(qw, scale, zp, wb, n8);
  }
  gemm_bf16_kernel<<<dim3(N / BN, M / BM), 256, 0, stream>>>(xb, wb, bias, out, M, N, K);
}
